// WalzeAttentionLayerV3_27358941675956
// MI455X (gfx1250) — matrix operand order repaired and checked by reading, not compiled, not run on hardware
//
#include <hip/hip_runtime.h>
#include <hip/hip_bf16.h>
#include <math.h>

// ---------------- constants for WalzeAttentionLayerV3 ----------------
#define DIMC   256
#define NHEADS 4
#define HDIM   64
#define BSZ    16
#define HGRID  32
#define WGRID  32
#define SQ     1024          // HGRID*WGRID
#define NTOK   (BSZ * SQ)    // 16384
#define EPSF   1e-5f
#define HALFW  16            // WINDOW/2

typedef _Float16 v16h __attribute__((ext_vector_type(16)));
typedef _Float16 v8h  __attribute__((ext_vector_type(8)));
typedef float    v8f  __attribute__((ext_vector_type(8)));

union AFrag { v16h v; v8h h[2]; };

__device__ __forceinline__ float gelu_exact(float x) {
    return 0.5f * x * (1.0f + erff(x * 0.70710678118654752440f));
}

// ---------------- K0: f32 -> f16 cast (weights) ----------------
__global__ __launch_bounds__(256)
void cast_f16_kernel(const float* __restrict__ in, _Float16* __restrict__ out, int n) {
    int i = blockIdx.x * 256 + threadIdx.x;
    if (i < n) out[i] = (_Float16)in[i];
}

// ---------------- K1: circular depthwise conv + BN + GELU + LN1 ----------------
// one block per token; 256 threads = channels
__global__ __launch_bounds__(256)
void walze_ln1_kernel(const float* __restrict__ x_grid,
                      const float* __restrict__ conv_w, const float* __restrict__ conv_b,
                      const float* __restrict__ bn_g,   const float* __restrict__ bn_b,
                      const float* __restrict__ bn_mean,const float* __restrict__ bn_var,
                      const float* __restrict__ ln1_g,  const float* __restrict__ ln1_b,
                      float* __restrict__ wn_f32,  _Float16* __restrict__ wn_f16,
                      _Float16* __restrict__ det_in)
{
    __shared__ float red[256];
    const int tok = blockIdx.x;
    const int c   = threadIdx.x;
    const int b   = tok >> 10;
    const int s   = tok & 1023;
    const int y   = s >> 5;
    const int x   = s & 31;

    const float* xc = x_grid + ((size_t)(b * DIMC + c)) * SQ;

    float acc = 0.0f;
    #pragma unroll
    for (int ky = 0; ky < 3; ++ky) {
        const int yy = (y + ky + 31) & 31;           // circular pad
        #pragma unroll
        for (int kx = 0; kx < 3; ++kx) {
            const int xx = (x + kx + 31) & 31;
            acc += xc[yy * WGRID + xx] * conv_w[c * 9 + ky * 3 + kx];
        }
    }
    const float center = xc[y * WGRID + x];
    const float pre    = acc + conv_b[c] + center;
    const float bn     = (pre - bn_mean[c]) * rsqrtf(bn_var[c] + EPSF) * bn_g[c] + bn_b[c];
    const float w      = gelu_exact(bn);

    // layernorm over 256 channels
    red[c] = w; __syncthreads();
    #pragma unroll
    for (int off = 128; off > 0; off >>= 1) { if (c < off) red[c] += red[c + off]; __syncthreads(); }
    const float mu = red[0] * (1.0f / 256.0f);
    __syncthreads();
    const float d0 = w - mu;
    red[c] = d0 * d0; __syncthreads();
    #pragma unroll
    for (int off = 128; off > 0; off >>= 1) { if (c < off) red[c] += red[c + off]; __syncthreads(); }
    const float var = red[0] * (1.0f / 256.0f);
    const float wn  = d0 * rsqrtf(var + EPSF) * ln1_g[c] + ln1_b[c];

    const size_t base = (size_t)tok * DIMC + c;
    wn_f32[base] = wn;
    wn_f16[base] = (_Float16)wn;
    det_in[(size_t)tok * (2 * DIMC) + c]        = (_Float16)center;  // x_before
    det_in[(size_t)tok * (2 * DIMC) + DIMC + c] = (_Float16)w;       // walze (pre-LN)
}

// ---------------- K2: generic WMMA GEMM  out[M,N] = A[M,K] * W[N,K]^T + bias ----------------
// each wave: one 16-row M tile x four 16-col N tiles (A-fragment reuse)
__global__ __launch_bounds__(256)
void gemm_wmma_f16_kernel(const _Float16* __restrict__ A,   // [M,K] row-major f16
                          const _Float16* __restrict__ W,   // [N,K] row-major f16
                          const float*    __restrict__ bias,// [N] or nullptr
                          float*          __restrict__ out, // [M,N] f32
                          int M, int N, int K)
{
    const int wave    = (blockIdx.x * 256 + threadIdx.x) >> 5;
    const int lane    = threadIdx.x & 31;
    const int ngroups = N >> 6;                 // groups of 64 output cols
    const int mt      = wave / ngroups;
    const int ng      = wave % ngroups;
    if (mt * 16 >= M) return;

    const int hi   = (lane >= 16);
    const int asel = hi ? 16 : 0;               // A: contiguous K 0-15 | 16-31
    const int bsel = hi ? 16 : 0;               // B: contiguous K 0-15 | 16-31
    const int mrow = mt * 16 + (lane & 15);
    const int n0   = ng * 64 + (lane & 15);

    const _Float16* arow = A + (size_t)mrow * K;
    const _Float16* wr0  = W + (size_t)(n0)      * K;
    const _Float16* wr1  = W + (size_t)(n0 + 16) * K;
    const _Float16* wr2  = W + (size_t)(n0 + 32) * K;
    const _Float16* wr3  = W + (size_t)(n0 + 48) * K;

    v8f acc0 = {}, acc1 = {}, acc2 = {}, acc3 = {};

    for (int kk = 0; kk < K; kk += 32) {
        AFrag a;
        a.h[0] = *(const v8h*)(arow + kk + asel);
        a.h[1] = *(const v8h*)(arow + kk + asel + 8);
        AFrag b0, b1, b2, b3;
        b0.h[0] = *(const v8h*)(wr0 + kk + bsel);
        b0.h[1] = *(const v8h*)(wr0 + kk + bsel + 8);
        b1.h[0] = *(const v8h*)(wr1 + kk + bsel);
        b1.h[1] = *(const v8h*)(wr1 + kk + bsel + 8);
        b2.h[0] = *(const v8h*)(wr2 + kk + bsel);
        b2.h[1] = *(const v8h*)(wr2 + kk + bsel + 8);
        b3.h[0] = *(const v8h*)(wr3 + kk + bsel);
        b3.h[1] = *(const v8h*)(wr3 + kk + bsel + 8);

        acc0 = __builtin_amdgcn_wmma_f32_16x16x32_f16(false, a.v, false, b0.v, (short)0, acc0, false, false);
        acc1 = __builtin_amdgcn_wmma_f32_16x16x32_f16(false, a.v, false, b1.v, (short)0, acc1, false, false);
        acc2 = __builtin_amdgcn_wmma_f32_16x16x32_f16(false, a.v, false, b2.v, (short)0, acc2, false, false);
        acc3 = __builtin_amdgcn_wmma_f32_16x16x32_f16(false, a.v, false, b3.v, (short)0, acc3, false, false);
    }

    // D layout: VGPR r, lane l -> M = mt*16 + r + (l>=16 ? 8:0), Ncol = lane&15
    const int mbase = mt * 16 + (hi ? 8 : 0);
    const float bv0 = bias ? bias[n0]      : 0.0f;
    const float bv1 = bias ? bias[n0 + 16] : 0.0f;
    const float bv2 = bias ? bias[n0 + 32] : 0.0f;
    const float bv3 = bias ? bias[n0 + 48] : 0.0f;
    #pragma unroll
    for (int r = 0; r < 8; ++r) {
        const size_t m = (size_t)(mbase + r);
        out[m * N + n0]      = acc0[r] + bv0;
        out[m * N + n0 + 16] = acc1[r] + bv1;
        out[m * N + n0 + 32] = acc2[r] + bv2;
        out[m * N + n0 + 48] = acc3[r] + bv3;
    }
}

// ---------------- K3: windowed attention (online softmax), one thread per (token, head) ----------------
__global__ __launch_bounds__(256)
void window_attn_kernel(const float* __restrict__ qkv,     // [NTOK, 768]
                        _Float16* __restrict__ ctx_f16)    // [NTOK, 256]
{
    const int gid = blockIdx.x * 256 + threadIdx.x;        // 0 .. NTOK*NHEADS-1
    const int tok = gid >> 2;
    const int h   = gid & 3;
    const int b   = tok >> 10;
    const int s   = tok & 1023;

    const float* qp = qkv + (size_t)tok * 768 + h * HDIM;
    float q[HDIM];
    #pragma unroll
    for (int d = 0; d < HDIM; ++d) q[d] = qp[d];

    float m = -1e30f, l = 0.0f;
    float ctx[HDIM];
    #pragma unroll
    for (int d = 0; d < HDIM; ++d) ctx[d] = 0.0f;

    #pragma unroll 1
    for (int ji = 0; ji <= 2 * HALFW; ++ji) {
        const int j = s - HALFW + ji;
        if (j < 0 || j >= SQ) continue;
        const float* kp = qkv + ((size_t)(b * SQ + j)) * 768 + DIMC + h * HDIM;
        float dot = 0.0f;
        #pragma unroll
        for (int d = 0; d < HDIM; ++d) dot += q[d] * kp[d];
        dot *= 0.125f;                                     // 1/sqrt(64)
        const float newm  = fmaxf(m, dot);
        const float scale = __expf(m - newm);
        const float p     = __expf(dot - newm);
        l = l * scale + p;
        const float* vp = qkv + ((size_t)(b * SQ + j)) * 768 + 2 * DIMC + h * HDIM;
        #pragma unroll
        for (int d = 0; d < HDIM; ++d) ctx[d] = ctx[d] * scale + p * vp[d];
        m = newm;
    }
    const float inv = 1.0f / l;
    _Float16* cp = ctx_f16 + (size_t)tok * DIMC + h * HDIM;
    #pragma unroll
    for (int d = 0; d < HDIM; ++d) cp[d] = (_Float16)(ctx[d] * inv);
}

// ---------------- K4: detector tail + gating + LN2 + residual ----------------
// one block per token; 256 threads = channels
__global__ __launch_bounds__(256)
void gate_ln2_kernel(const float* __restrict__ x_grid,
                     const float* __restrict__ h1,       // [NTOK, 64] raw GEMM out
                     const float* __restrict__ d1_b,     // [64]
                     const float* __restrict__ d2_w,     // [64]
                     const float* __restrict__ d2_b,     // [1]
                     const float* __restrict__ wn_f32,   // [NTOK, 256]
                     const float* __restrict__ attn_out, // [NTOK, 256]
                     const float* __restrict__ ln2_g, const float* __restrict__ ln2_b,
                     float* __restrict__ out)
{
    __shared__ float red[256];
    __shared__ float spike_sh;
    const int tok = blockIdx.x;
    const int c   = threadIdx.x;
    const int b   = tok >> 10;
    const int s   = tok & 1023;

    // spike detector: gelu(h1 + d1_b) . d2_w + d2_b -> sigmoid
    float p = 0.0f;
    if (c < 64) {
        const float hv = h1[(size_t)tok * 64 + c] + d1_b[c];
        p = gelu_exact(hv) * d2_w[c];
    }
    red[c] = p; __syncthreads();
    #pragma unroll
    for (int off = 128; off > 0; off >>= 1) { if (c < off) red[c] += red[c + off]; __syncthreads(); }
    if (c == 0) {
        const float logit = red[0] + d2_b[0];
        spike_sh = 1.0f / (1.0f + __expf(-logit));
    }
    __syncthreads();
    const float spike = spike_sh;

    const size_t base = (size_t)tok * DIMC + c;
    const float g = (1.0f - spike) * wn_f32[base] + spike * attn_out[base];

    // layernorm 2
    red[c] = g; __syncthreads();
    #pragma unroll
    for (int off = 128; off > 0; off >>= 1) { if (c < off) red[c] += red[c + off]; __syncthreads(); }
    const float mu = red[0] * (1.0f / 256.0f);
    __syncthreads();
    const float d0 = g - mu;
    red[c] = d0 * d0; __syncthreads();
    #pragma unroll
    for (int off = 128; off > 0; off >>= 1) { if (c < off) red[c] += red[c + off]; __syncthreads(); }
    const float var = red[0] * (1.0f / 256.0f);
    const float fin = d0 * rsqrtf(var + EPSF) * ln2_g[c] + ln2_b[c];

    const size_t oidx = ((size_t)(b * DIMC + c)) * SQ + s;   // NCHW
    out[oidx] = x_grid[oidx] + fin;
}

// ---------------- host launcher ----------------
extern "C" void kernel_launch(void* const* d_in, const int* in_sizes, int n_in,
                              void* d_out, int out_size, void* d_ws, size_t ws_size,
                              hipStream_t stream) {
    (void)in_sizes; (void)n_in; (void)out_size; (void)ws_size;
    const float* x_grid  = (const float*)d_in[0];
    const float* conv_w  = (const float*)d_in[1];
    const float* conv_b  = (const float*)d_in[2];
    const float* bn_g    = (const float*)d_in[3];
    const float* bn_b    = (const float*)d_in[4];
    const float* bn_mean = (const float*)d_in[5];
    const float* bn_var  = (const float*)d_in[6];
    const float* ln1_g   = (const float*)d_in[7];
    const float* ln1_b   = (const float*)d_in[8];
    const float* ln2_g   = (const float*)d_in[9];
    const float* ln2_b   = (const float*)d_in[10];
    const float* in_w    = (const float*)d_in[11];
    const float* in_b    = (const float*)d_in[12];
    const float* out_w   = (const float*)d_in[13];
    const float* out_b   = (const float*)d_in[14];
    const float* d1_w    = (const float*)d_in[15];
    const float* d1_b    = (const float*)d_in[16];
    const float* d2_w    = (const float*)d_in[17];
    const float* d2_b    = (const float*)d_in[18];
    float* out = (float*)d_out;

    // workspace carve-out (16B+ aligned chunks)
    char* ws = (char*)d_ws;
    size_t off = 0;
    auto carve = [&](size_t bytes) -> void* {
        void* p = ws + off;
        off += (bytes + 255) & ~(size_t)255;
        return p;
    };
    float*     wn_f32   = (float*)    carve((size_t)NTOK * DIMC * 4);      // walze_seq_n f32
    _Float16*  wn_f16   = (_Float16*) carve((size_t)NTOK * DIMC * 2);      // walze_seq_n f16
    _Float16*  det_in   = (_Float16*) carve((size_t)NTOK * 2 * DIMC * 2);  // [x || walze] f16
    float*     qkv      = (float*)    carve((size_t)NTOK * 768 * 4);
    _Float16*  ctx_f16  = (_Float16*) carve((size_t)NTOK * DIMC * 2);
    float*     attn_out = (float*)    carve((size_t)NTOK * DIMC * 4);
    float*     h1       = (float*)    carve((size_t)NTOK * 64 * 4);
    _Float16*  in_w_h   = (_Float16*) carve((size_t)768 * DIMC * 2);
    _Float16*  out_w_h  = (_Float16*) carve((size_t)DIMC * DIMC * 2);
    _Float16*  d1_w_h   = (_Float16*) carve((size_t)64 * 2 * DIMC * 2);

    // K0: cast weights to f16
    cast_f16_kernel<<<(768 * DIMC + 255) / 256, 256, 0, stream>>>(in_w, in_w_h, 768 * DIMC);
    cast_f16_kernel<<<(DIMC * DIMC + 255) / 256, 256, 0, stream>>>(out_w, out_w_h, DIMC * DIMC);
    cast_f16_kernel<<<(64 * 2 * DIMC + 255) / 256, 256, 0, stream>>>(d1_w, d1_w_h, 64 * 2 * DIMC);

    // K1: conv + BN + GELU + LN1 (+ detector input staging)
    walze_ln1_kernel<<<NTOK, 256, 0, stream>>>(x_grid, conv_w, conv_b, bn_g, bn_b,
                                               bn_mean, bn_var, ln1_g, ln1_b,
                                               wn_f32, wn_f16, det_in);

    // K2a: QKV GEMM  [16384,256] x [768,256]^T -> [16384,768]
    {
        const int waves = (NTOK / 16) * (768 / 64);   // 12288
        gemm_wmma_f16_kernel<<<waves / 8, 256, 0, stream>>>(wn_f16, in_w_h, in_b, qkv,
                                                            NTOK, 768, DIMC);
    }

    // K3: windowed attention -> ctx (f16)
    window_attn_kernel<<<(NTOK * NHEADS) / 256, 256, 0, stream>>>(qkv, ctx_f16);

    // K2b: out-proj GEMM  [16384,256] x [256,256]^T -> attn_out
    {
        const int waves = (NTOK / 16) * (DIMC / 64);  // 4096
        gemm_wmma_f16_kernel<<<waves / 8, 256, 0, stream>>>(ctx_f16, out_w_h, out_b, attn_out,
                                                            NTOK, DIMC, DIMC);
    }

    // K2c: detector L1 GEMM  [16384,512] x [64,512]^T -> h1 (bias applied later with GELU)
    {
        const int waves = (NTOK / 16) * (64 / 64);    // 1024
        gemm_wmma_f16_kernel<<<waves / 8, 256, 0, stream>>>(det_in, d1_w_h, nullptr, h1,
                                                            NTOK, 64, 2 * DIMC);
    }

    // K4: detector tail + gating + LN2 + residual
    gate_ln2_kernel<<<NTOK, 256, 0, stream>>>(x_grid, h1, d1_b, d2_w, d2_b,
                                              wn_f32, attn_out, ln2_g, ln2_b, out);
}
